// IFFN_54262616817909
// MI455X (gfx1250) — matrix operand order repaired and checked by reading, not compiled, not run on hardware
//
#include <hip/hip_runtime.h>
#include <hip/hip_bf16.h>
#include <stdint.h>

// MoE SwiGLU FFN for MI455X (gfx1250, wave32, WMMA bf16 16x16x32).
// B=2,T=1024 -> N=2048 tokens; D=2048; H=1408; 16 experts, top-4 routing.
// Gather-based routing: only top-4 experts computed per token (4x less work
// than the dense reference, identical math since masked probs are zero).
// Weights stream from HBM exactly once per 16-token tile; activations stay
// in LDS. Memory-bound: ~590 MB weights @ 23.3 TB/s ~= 27 us floor.

#define NTOK   2048
#define DDIM   2048
#define HDIM   1408
#define NEXP   16
#define TOPK   4
#define TILE_M 16
#define CHUNK  256      // X staging chunk (columns)

typedef __attribute__((ext_vector_type(16))) __bf16 v16bf;
typedef __attribute__((ext_vector_type(8)))  float  v8f;
typedef int v4i_vs __attribute__((vector_size(16)));   // matches builtin param

#if defined(__has_builtin)
#if __has_builtin(__builtin_amdgcn_global_load_async_to_lds_b128)
#define HAVE_ASYNC_B128 1
#endif
#if __has_builtin(__builtin_amdgcn_s_wait_asynccnt)
#define HAVE_WAIT_ASYNC 1
#endif
#endif

// native bf16 convert (v_cvt_*bf16_f32 on gfx1250)
__device__ __forceinline__ __bf16 f2bf(float f) { return (__bf16)f; }

// sigmoid via v_rcp_f32 instead of IEEE divide chains
__device__ __forceinline__ float fsig(float t) {
  return __builtin_amdgcn_rcpf(1.f + __expf(-t));
}

// async global->LDS copy of 16 bytes (per-lane addresses), ASYNCcnt-tracked.
// LDS pointer truncation to 32 bits matches the ISA LDS-aperture rule.
__device__ __forceinline__ void async_copy16(const float* g, float* l) {
#if HAVE_ASYNC_B128
  __builtin_amdgcn_global_load_async_to_lds_b128(
      (__attribute__((address_space(1))) v4i_vs*)(uintptr_t)g,
      (__attribute__((address_space(3))) v4i_vs*)(uint32_t)(uintptr_t)l, 0, 0);
#else
  *(float4*)l = *(const float4*)g;
#endif
}

__device__ __forceinline__ void wait_async0() {
#if HAVE_WAIT_ASYNC
  __builtin_amdgcn_s_wait_asynccnt(0);
#else
  asm volatile("s_wait_asynccnt 0x0" ::: "memory");
#endif
}

union AFrag { v16bf v; uint4 q[2]; };

// A fragment: 16xK row-major bf16 tile in LDS, row stride ldk (elements).
// ISA 7.12.2: lanes 0-15 -> M=lane, K = k0+{0..7,16..23}; lanes 16-31 shift K by 8.
__device__ __forceinline__ v16bf load_a(const __bf16* base, int ldk, int k0) {
  int lane = threadIdx.x & 31;
  int m  = lane & 15;
  int kb = (lane < 16) ? 0 : 8;
  const __bf16* p = base + m * ldk + k0 + kb;   // 16B aligned
  AFrag f;
  f.q[0] = *(const uint4*)(p);
  f.q[1] = *(const uint4*)(p + 16);
  return f.v;
}

// B fragment: KxN fp32 global weight tile, row stride ldn, converted to bf16.
// lanes 0-15 -> N=lane, K = k0+{0..7,16..23}; lanes 16-31 shift K by 8.
__device__ __forceinline__ v16bf load_b(const float* __restrict__ w, int ldn,
                                        int k0, int n0) {
  int lane = threadIdx.x & 31;
  int n  = lane & 15;
  int kb = (lane < 16) ? 0 : 8;
  const float* p = w + (size_t)(k0 + kb) * ldn + n0 + n;
  v16bf b;
#pragma unroll
  for (int i = 0; i < 16; ++i) b[i] = f2bf(p[(size_t)(i + (i & 8)) * ldn]);
  return b;
}

__global__ void moe_zero(int* cnt) {
  if (threadIdx.x < NEXP) cnt[threadIdx.x] = 0;
}

// One wave32 per token: logits, sigmoid+bias, top-4, renormalize, append to
// per-expert gather lists.
__global__ void moe_gate(const float* __restrict__ x,
                         const float* __restrict__ gw,
                         const float* __restrict__ gb,
                         float* __restrict__ probs,
                         int* __restrict__ cnt,
                         int* __restrict__ lists) {
  int lane = threadIdx.x & 31;
  int n = blockIdx.x * (blockDim.x >> 5) + (threadIdx.x >> 5);
  if (n >= NTOK) return;
  const float* xr = x + (size_t)n * DDIM;

  float myscore = 0.f;
  for (int e = 0; e < NEXP; ++e) {
    const float* wr = gw + e * DDIM;
    float acc = 0.f;
    for (int d = lane; d < DDIM; d += 32) acc += wr[d] * xr[d];
    for (int off = 16; off; off >>= 1) acc += __shfl_xor(acc, off, 32);
    if (lane == e) myscore = acc;
  }
  // scores = sigmoid(logits) + bias (reference order)
  float sc = (lane < NEXP) ? (fsig(myscore) + gb[lane]) : -1e30f;

  float cur = sc, wsum = 0.f, mysel = 0.f;
  bool sel = false;
  for (int t = 0; t < TOPK; ++t) {
    float s = cur; int idx = lane;
    for (int off = 16; off; off >>= 1) {
      float os = __shfl_xor(s, off, 32);
      int   oi = __shfl_xor(idx, off, 32);
      if (os > s || (os == s && oi < idx)) { s = os; idx = oi; }
    }
    wsum += s;
    if (lane == idx) { sel = true; mysel = sc; cur = -1e30f; }
  }
  if (lane < NEXP) {
    probs[(size_t)n * NEXP + lane] = sel ? (mysel * __builtin_amdgcn_rcpf(wsum)) : 0.f;
    if (sel) {
      int pos = atomicAdd(&cnt[lane], 1);
      lists[lane * NTOK + pos] = n;
    }
  }
}

// One workgroup = one (expert, 16-token tile). 8 waves.
// Stage A: h[16,H] = epilogue(x16 @ W1, x16 @ W3) into LDS (bf16).
// Stage B: out16 = h @ W2, scaled by router prob, atomic-add (routed) or
//          plain store (shared).
template <int ROUTED>
__global__ void moe_ffn_tile(const float* __restrict__ x,
                             const float* __restrict__ w1,
                             const float* __restrict__ w3,
                             const float* __restrict__ w2,
                             const float* __restrict__ probs,
                             const int* __restrict__ cnt,
                             const int* __restrict__ lists,
                             float* __restrict__ out) {
  extern __shared__ char smem[];
  __bf16* xt  = (__bf16*)smem;                                  // 16 x 2048 bf16 (64KB)
  __bf16* ht  = (__bf16*)(smem + TILE_M * DDIM * 2);            // 16 x 1408 bf16 (44KB)
  float*  xst = (float*)(smem + TILE_M * DDIM * 2 + TILE_M * HDIM * 2); // 16x256 f32 (16KB)
  int*    tks = (int*)(xst + TILE_M * CHUNK);
  float*  pts = (float*)(tks + TILE_M);

  int tile = blockIdx.x;
  int e    = blockIdx.y;
  int nvalid = TILE_M;
  const float *pw1 = w1, *pw3 = w3, *pw2 = w2;
  if (ROUTED) {
    int c = cnt[e];
    if (tile * TILE_M >= c) return;                 // uniform early-out
    nvalid = min(TILE_M, c - tile * TILE_M);
    size_t eb = (size_t)e * DDIM * HDIM;
    pw1 = w1 + eb; pw3 = w3 + eb;
    pw2 = w2 + (size_t)e * HDIM * DDIM;
  }
  int tid = threadIdx.x;
  if (tid < TILE_M) {
    int tok; float p;
    if (ROUTED) {
      if (tid < nvalid) {
        tok = lists[e * NTOK + tile * TILE_M + tid];
        p = probs[(size_t)tok * NEXP + e];
      } else { tok = 0; p = 0.f; }
    } else { tok = tile * TILE_M + tid; p = 1.f; }
    tks[tid] = tok; pts[tid] = p;
  }
  __syncthreads();

  // ---- stage X tile: async gather fp32 chunks -> LDS, convert to bf16 ----
  for (int c0 = 0; c0 < DDIM; c0 += CHUNK) {
#pragma unroll
    for (int it = 0; it < (TILE_M * CHUNK) / 4 / 256; ++it) {   // 4 iters
      int i = (tid + it * 256) * 4;                // 16B granules
      int m = i >> 8;                              // / CHUNK
      int k = i & (CHUNK - 1);
      async_copy16(x + (size_t)tks[m] * DDIM + c0 + k, xst + i);
    }
    wait_async0();
    __syncthreads();
    // fp32 -> bf16, packed pair stores (4B ds_store per 2 elements)
    for (int i = tid * 2; i < TILE_M * CHUNK; i += 512) {
      int m = i >> 8;
      int k = i & (CHUNK - 1);
      float2 f = *(const float2*)(xst + i);
      union { __bf16 h[2]; uint32_t u; } pk;
      pk.h[0] = f2bf(f.x); pk.h[1] = f2bf(f.y);
      *(uint32_t*)(xt + m * DDIM + c0 + k) = pk.u;
    }
    __syncthreads();
  }

  int wave = tid >> 5;
  const int NW = 8;
  int lane = tid & 31;
  int col  = lane & 15;
  int mofs = (lane < 16) ? 0 : 8;

  // ---- Stage A: h = f(x@W1, x@W3), H = 88 N-tiles of 16 ----
  for (int nt = wave; nt < HDIM / 16; nt += NW) {
    int n0 = nt * 16;
    v8f a1 = {}; v8f a3 = {};
    for (int k0 = 0; k0 < DDIM; k0 += 32) {
      __builtin_prefetch(pw1 + (size_t)(k0 + 32) * HDIM + n0, 0, 1);
      __builtin_prefetch(pw3 + (size_t)(k0 + 32) * HDIM + n0, 0, 1);
      v16bf af = load_a(xt, DDIM, k0);
      v16bf b1 = load_b(pw1, HDIM, k0, n0);
      v16bf b3 = load_b(pw3, HDIM, k0, n0);
      a1 = __builtin_amdgcn_wmma_f32_16x16x32_bf16(false, af, false, b1,
                                                   (short)0, a1, false, false);
      a3 = __builtin_amdgcn_wmma_f32_16x16x32_bf16(false, af, false, b3,
                                                   (short)0, a3, false, false);
    }
#pragma unroll
    for (int v = 0; v < 8; ++v) {
      float h;
      if (ROUTED) {            // silu(h1 * h2)  (as in reference source)
        float t = a1[v] * a3[v];
        h = t * fsig(t);
      } else {                 // silu(x@W1) * (x@W3)
        h = a1[v] * fsig(a1[v]) * a3[v];
      }
      ht[(mofs + v) * HDIM + n0 + col] = f2bf(h);
    }
  }
  __syncthreads();

  // ---- Stage B: out = h @ W2, D = 128 N-tiles of 16 ----
  for (int nt = wave; nt < DDIM / 16; nt += NW) {
    int n0 = nt * 16;
    v8f acc = {};
    for (int k0 = 0; k0 < HDIM; k0 += 32) {
      __builtin_prefetch(pw2 + (size_t)(k0 + 32) * DDIM + n0, 0, 1);
      v16bf af = load_a(ht, HDIM, k0);
      v16bf bf = load_b(pw2, DDIM, k0, n0);
      acc = __builtin_amdgcn_wmma_f32_16x16x32_bf16(false, af, false, bf,
                                                    (short)0, acc, false, false);
    }
#pragma unroll
    for (int v = 0; v < 8; ++v) {
      int m = mofs + v;
      float val = acc[v] * pts[m];
      float* dst = &out[(size_t)tks[m] * DDIM + n0 + col];
      if (ROUTED) {
        if (m < nvalid) unsafeAtomicAdd(dst, val);   // global_atomic_add_f32
      } else {
        *dst = val;                                  // initializes d_out
      }
    }
  }
}

extern "C" void kernel_launch(void* const* d_in, const int* in_sizes, int n_in,
                              void* d_out, int out_size, void* d_ws, size_t ws_size,
                              hipStream_t stream) {
  const float* x   = (const float*)d_in[0];
  const float* gw  = (const float*)d_in[1];
  const float* gb  = (const float*)d_in[2];
  const float* w1s = (const float*)d_in[3];
  const float* w2s = (const float*)d_in[4];
  const float* w3s = (const float*)d_in[5];
  const float* w1r = (const float*)d_in[6];
  const float* w2r = (const float*)d_in[7];
  const float* w3r = (const float*)d_in[8];
  float* out = (float*)d_out;

  // workspace: probs[N,NE] | cnt[NE](pad 32) | lists[NE,N]   (~256 KB)
  char*  ws    = (char*)d_ws;
  float* probs = (float*)ws;
  int*   cnt   = (int*)(ws + (size_t)NTOK * NEXP * 4);
  int*   lists = cnt + 32;

  moe_zero<<<1, 32, 0, stream>>>(cnt);
  moe_gate<<<NTOK / 8, 256, 0, stream>>>(x, gw, gb, probs, cnt, lists);

  size_t shmem = (size_t)TILE_M * DDIM * 2     // x tile bf16
               + (size_t)TILE_M * HDIM * 2     // h tile bf16
               + (size_t)TILE_M * CHUNK * 4    // fp32 staging chunk
               + TILE_M * 4 + TILE_M * 4;      // token ids + probs
  dim3 gs(NTOK / TILE_M, 1);
  moe_ffn_tile<0><<<gs, 256, shmem, stream>>>(x, w1s, w3s, w2s, probs, cnt, lists, out);
  dim3 gr(NTOK / TILE_M, NEXP);
  moe_ffn_tile<1><<<gr, 256, shmem, stream>>>(x, w1r, w3r, w2r, probs, cnt, lists, out);
}
